// CaptionModel_3882650436479
// MI455X (gfx1250) — matrix operand order repaired and checked by reading, not compiled, not run on hardware
//
#include <hip/hip_runtime.h>
#include <stdint.h>

typedef __attribute__((ext_vector_type(16))) __bf16 v16bf;
typedef __attribute__((ext_vector_type(8)))  float  v8f;

namespace {
constexpr int B_ = 16, R_ = 100, V_ = 50000, D_ = 1024, T_ = 20;
constexpr int VP1 = V_ + 1;          // 50001
constexpr int RP1 = R_ + 1;          // 101
constexpr int VT  = VP1 + R_;        // 50101 merged vocab
constexpr int NPAD = 50016;          // 3126 * 16
constexpr int NT_WL = NPAD / 16;     // 3126 N-tiles
constexpr int KT_WL = D_ / 32;       // 32 K-tiles
constexpr int K3 = 3 * D_;           // 3072 (xt|h|ctx)
constexpr int KT_X = K3 / 32;        // 96
constexpr int NT_X = D_ / 16;        // 64
constexpr int NCAND = B_ * VT;       // 801616
constexpr int CH = 8192;
constexpr int NB1 = (NCAND + CH - 1) / CH;   // 98
constexpr float NEGL = -1e10f;
constexpr float NEGBIG = -3.0e38f;

constexpr size_t alignup(size_t x) { return (x + 255) & ~(size_t)255; }
constexpr size_t OFF_WLP = 0;                                              // bf16 Wl pack 102.4MB
constexpr size_t OFF_XP  = alignup(OFF_WLP + (size_t)NT_WL * KT_WL * 1024);// bf16 [Wx;Wh;Wc] pack
constexpr size_t OFF_APW = alignup(OFF_XP  + (size_t)NT_X * KT_X * 1024);  // A frags for Wl gemm
constexpr size_t OFF_APX = alignup(OFF_APW + (size_t)KT_WL * 1024);
constexpr size_t OFF_LOG = alignup(OFF_APX + (size_t)KT_X * 1024);         // logits 16 x NPAD f32
constexpr size_t OFF_LZ  = alignup(OFF_LOG + (size_t)B_ * NPAD * 4);
constexpr size_t OFF_DLP = alignup(OFF_LZ  + 64);
constexpr size_t OFF_DLG = alignup(OFF_DLP + B_ * RP1 * 4);
constexpr size_t OFF_MSK = alignup(OFF_DLG + B_ * RP1 * 4);
constexpr size_t OFF_RNN = alignup(OFF_MSK + B_ * RP1 * 4);
constexpr size_t OFF_HST = alignup(OFF_RNN + B_ * D_ * 4);
constexpr size_t OFF_XT  = alignup(OFF_HST + B_ * D_ * 4);
constexpr size_t OFF_CTX = alignup(OFF_XT  + B_ * D_ * 4);
constexpr size_t OFF_LPS = alignup(OFF_CTX + B_ * D_ * 4);
constexpr size_t OFF_SEQ = alignup(OFF_LPS + 64);
constexpr size_t OFF_LPB = alignup(OFF_SEQ + T_ * B_ * 4);
constexpr size_t OFF_PV  = alignup(OFF_LPB + T_ * B_ * 4);
constexpr size_t OFF_PI  = alignup(OFF_PV  + NB1 * 16 * 4);
constexpr size_t OFF_SEL = alignup(OFF_PI  + NB1 * 16 * 4);
} // namespace

// round-to-nearest-even fp32 -> bf16 bits (bit trick, avoids scalar __bf16 math)
__device__ __forceinline__ uint16_t f2bf(float f) {
  uint32_t u = __builtin_bit_cast(uint32_t, f);
  u += 0x7fffu + ((u >> 16) & 1u);
  return (uint16_t)(u >> 16);
}

// ---------------------------------------------------------------------------
// One-time pre-pack of Wl (fp32 [D, VP1]) into bf16 WMMA-B fragment layout.
// Fragment f = ntile*KT_WL + ktile holds a 32x16 (KxN) tile; within fragment:
//   lane L in [0,16): col n=L,   elems e -> K = e        (K 0..15)
//   lane L in [16,32): col n=L-16, elems e -> K = 16 + e (K 16..31)
// stored at pack[f*512 + L*16 + e]  (each lane's 16 bf16 contiguous = 32B).
__global__ __launch_bounds__(256) void pack_wl(const float* __restrict__ Wl,
                                               uint16_t* __restrict__ pack) {
  size_t tid = (size_t)blockIdx.x * 256 + threadIdx.x;
  if (tid >= (size_t)D_ * NPAD) return;
  int k = (int)(tid / NPAD);
  int n = (int)(tid % NPAD);
  int kt = k >> 5, kk = k & 31, nt = n >> 4;
  int L = (n & 15) + ((kk >> 4) << 4);
  int e = kk & 15;
  float v = (n < VP1) ? Wl[(size_t)k * VP1 + n] : 0.0f;
  pack[(size_t)(nt * KT_WL + kt) * 512 + L * 16 + e] = f2bf(v);
}

// One-time pre-pack of [Wx;Wh;Wc] (3072x1024) into the same B-fragment layout.
__global__ __launch_bounds__(256) void pack_wxhc(const float* __restrict__ Wx,
                                                 const float* __restrict__ Wh,
                                                 const float* __restrict__ Wc,
                                                 uint16_t* __restrict__ pack) {
  size_t tid = (size_t)blockIdx.x * 256 + threadIdx.x;
  if (tid >= (size_t)K3 * D_) return;
  int k = (int)(tid / D_);
  int n = (int)(tid % D_);
  float v = (k < D_) ? Wx[(size_t)k * D_ + n]
          : (k < 2 * D_) ? Wh[(size_t)(k - D_) * D_ + n]
                         : Wc[(size_t)(k - 2 * D_) * D_ + n];
  int kt = k >> 5, kk = k & 31, nt = n >> 4;
  int L = (n & 15) + ((kk >> 4) << 4);
  int e = kk & 15;
  pack[(size_t)(nt * KT_X + kt) * 512 + L * 16 + e] = f2bf(v);
}

// ---------------------------------------------------------------------------
// Pack a [16, K] fp32 activation into bf16 WMMA-A fragments in contiguous k order (same as the B pack):
//   lane = m + 16*(kk>>4) ; elem = kk&15
__device__ __forceinline__ void packA_one(int m, int k, float val, int ktiles,
                                          uint16_t* pack) {
  int kt = k >> 5, kk = k & 31;
  int lane = m + ((kk >> 4) << 4);
  int e = kk & 15;
  pack[(size_t)kt * 512 + lane * 16 + e] = f2bf(val);
}

__global__ __launch_bounds__(256) void packA_wl(const float* __restrict__ rnn,
                                                uint16_t* __restrict__ pack) {
  int tid = blockIdx.x * 256 + threadIdx.x;
  if (tid >= B_ * D_) return;
  int m = tid / D_, k = tid % D_;
  packA_one(m, k, rnn[(size_t)m * D_ + k], KT_WL, pack);
}

__global__ __launch_bounds__(256) void packA_x(const float* __restrict__ xt,
                                               const float* __restrict__ hst,
                                               const float* __restrict__ ctx,
                                               uint16_t* __restrict__ pack) {
  int tid = blockIdx.x * 256 + threadIdx.x;
  if (tid >= B_ * K3) return;
  int m = tid / K3, k = tid % K3;
  float v = (k < D_) ? xt[(size_t)m * D_ + k]
          : (k < 2 * D_) ? hst[(size_t)m * D_ + (k - D_)]
                         : ctx[(size_t)m * D_ + (k - 2 * D_)];
  packA_one(m, k, v, KT_X, pack);
}

// ---------------------------------------------------------------------------
// Big GEMM: logits[16, NPAD] = rnn_out(bf16 A frags) x Wl(bf16 B frags).
// One wave per 16-col N-tile, 32 wmma along K. A (32KB) staged in LDS, shared
// by the 8 waves of the block.
__global__ __launch_bounds__(256) void gemm_wl(const uint16_t* __restrict__ Apack,
                                               const uint16_t* __restrict__ Bpack,
                                               float* __restrict__ logits) {
  __shared__ __attribute__((aligned(32))) uint16_t sA[KT_WL * 512];  // 32 KB
  const uint4* gA = (const uint4*)Apack;
  uint4* lA = (uint4*)sA;
  for (int i = threadIdx.x; i < (KT_WL * 512 * 2) / 16; i += 256) lA[i] = gA[i];
  __syncthreads();

  int wave = threadIdx.x >> 5, lane = threadIdx.x & 31;
  int nt = blockIdx.x * 8 + wave;
  if (nt >= NT_WL) return;

  v8f acc = {0.f, 0.f, 0.f, 0.f, 0.f, 0.f, 0.f, 0.f};
  const v16bf* ap = (const v16bf*)sA;
  const v16bf* bp = (const v16bf*)(Bpack + (size_t)nt * KT_WL * 512);
#pragma unroll 4
  for (int kt = 0; kt < KT_WL; ++kt) {
    v16bf a = ap[kt * 32 + lane];
    v16bf b = bp[kt * 32 + lane];
    if (kt + 1 < KT_WL) __builtin_prefetch(bp + (kt + 1) * 32 + lane, 0, 0);
    acc = __builtin_amdgcn_wmma_f32_16x16x32_bf16(false, a, false, b,
                                                  (short)0, acc, false, false);
  }
  // C layout: lanes 0-15 -> M=r, lanes 16-31 -> M=8+r ; N = lane&15
  int col = lane & 15, rbase = (lane >> 4) << 3;
  int n = nt * 16 + col;
#pragma unroll
  for (int r = 0; r < 8; ++r)
    logits[(size_t)(rbase + r) * NPAD + n] = acc[r];
}

// Fused recurrent GEMM: h_new = tanh([xt|h|ctx] @ [Wx;Wh;Wc]) -> rnn & hst.
__global__ __launch_bounds__(256) void gemm_x_tanh(const uint16_t* __restrict__ Apack,
                                                   const uint16_t* __restrict__ Bpack,
                                                   float* __restrict__ rnn,
                                                   float* __restrict__ hst) {
  int wave = threadIdx.x >> 5, lane = threadIdx.x & 31;
  int nt = blockIdx.x * 8 + wave;
  if (nt >= NT_X) return;
  v8f acc = {0.f, 0.f, 0.f, 0.f, 0.f, 0.f, 0.f, 0.f};
  const v16bf* ap = (const v16bf*)Apack;
  const v16bf* bp = (const v16bf*)(Bpack + (size_t)nt * KT_X * 512);
#pragma unroll 4
  for (int kt = 0; kt < KT_X; ++kt) {
    acc = __builtin_amdgcn_wmma_f32_16x16x32_bf16(false, ap[kt * 32 + lane],
                                                  false, bp[kt * 32 + lane],
                                                  (short)0, acc, false, false);
  }
  int col = lane & 15, rbase = (lane >> 4) << 3;
  int n = nt * 16 + col;
#pragma unroll
  for (int r = 0; r < 8; ++r) {
    float hv = tanhf(acc[r]);
    rnn[(size_t)(rbase + r) * D_ + n] = hv;
    hst[(size_t)(rbase + r) * D_ + n] = hv;
  }
}

// ---------------------------------------------------------------------------
// Row-wise logsumexp over the word logits (online max/sum, one block per row).
__global__ __launch_bounds__(256) void lse_k(const float* __restrict__ logits,
                                             float* __restrict__ logZ) {
  int b = blockIdx.x;
  float m = NEGBIG, s = 0.f;
  for (int i = threadIdx.x; i < VP1; i += 256) {
    float x = logits[(size_t)b * NPAD + i];
    if (x > m) { s = s * __expf(m - x) + 1.0f; m = x; }
    else       { s += __expf(x - m); }
  }
  __shared__ float sm[256], ss[256];
  sm[threadIdx.x] = m; ss[threadIdx.x] = s;
  __syncthreads();
  for (int o = 128; o; o >>= 1) {
    if (threadIdx.x < o) {
      float m1 = sm[threadIdx.x], s1 = ss[threadIdx.x];
      float m2 = sm[threadIdx.x + o], s2 = ss[threadIdx.x + o];
      float M = fmaxf(m1, m2);
      sm[threadIdx.x] = M;
      ss[threadIdx.x] = s1 * __expf(m1 - M) + s2 * __expf(m2 - M);
    }
    __syncthreads();
  }
  if (threadIdx.x == 0) logZ[b] = sm[0] + __logf(ss[0]);
}

// det_lp = log_softmax(det_lg) over [16,101] (tiny).
__global__ void det_lsm(const float* __restrict__ det_lg, float* __restrict__ det_lp) {
  int b = threadIdx.x;
  if (b >= B_) return;
  float m = NEGBIG;
  for (int i = 0; i < RP1; ++i) m = fmaxf(m, det_lg[b * RP1 + i]);
  float s = 0.f;
  for (int i = 0; i < RP1; ++i) s += __expf(det_lg[b * RP1 + i] - m);
  float lz = m + __logf(s);
  for (int i = 0; i < RP1; ++i) det_lp[b * RP1 + i] = det_lg[b * RP1 + i] - lz;
}

// ---------------------------------------------------------------------------
// top-k stage 1: each block builds cand values for an 8192-chunk in LDS and
// extracts its local top-16 via repeated argmax (tie -> smaller global index).
__global__ __launch_bounds__(256) void topk1(const float* __restrict__ logits,
                                             const float* __restrict__ logZ,
                                             const float* __restrict__ det_lp,
                                             const float* __restrict__ lp_sum,
                                             int t, float* __restrict__ pv,
                                             int* __restrict__ pi) {
  __shared__ float v[CH];           // 32 KB
  __shared__ float sv[256];
  __shared__ int   si[256];
  int base = blockIdx.x * CH;
  for (int i = threadIdx.x; i < CH; i += 256) {
    int idx = base + i;
    float val = NEGBIG;
    if (idx < NCAND) {
      int b = idx / VT, c = idx - b * VT;
      if (t == 0 && b > 0) val = NEGL;
      else if (c < VP1)
        val = lp_sum[b] + logits[(size_t)b * NPAD + c] - logZ[b] + det_lp[b * RP1];
      else
        val = lp_sum[b] + det_lp[b * RP1 + (c - V_)];
    }
    v[i] = val;
  }
  __syncthreads();
  for (int it = 0; it < 16; ++it) {
    int bs = threadIdx.x;
    float bv = v[bs];
    for (int i = threadIdx.x + 256; i < CH; i += 256) {
      float x = v[i];
      if (x > bv) { bv = x; bs = i; }
    }
    sv[threadIdx.x] = bv; si[threadIdx.x] = bs;
    __syncthreads();
    for (int o = 128; o; o >>= 1) {
      if (threadIdx.x < o) {
        float x = sv[threadIdx.x + o]; int s2 = si[threadIdx.x + o];
        if (x > sv[threadIdx.x] ||
            (x == sv[threadIdx.x] && s2 < si[threadIdx.x])) {
          sv[threadIdx.x] = x; si[threadIdx.x] = s2;
        }
      }
      __syncthreads();
    }
    if (threadIdx.x == 0) {
      pv[blockIdx.x * 16 + it] = sv[0];
      pi[blockIdx.x * 16 + it] = base + si[0];
      v[si[0]] = NEGBIG;
    }
    __syncthreads();
  }
}

// top-k stage 2 + full beam-state update (single block).
__global__ __launch_bounds__(256) void topk2_update(
    const float* __restrict__ pv, const int* __restrict__ pi,
    const float* __restrict__ logits, const float* __restrict__ logZ,
    const float* __restrict__ det_lp, float* __restrict__ lp_sum,
    int* __restrict__ seq, float* __restrict__ lpb, float* __restrict__ hst,
    float* __restrict__ mask, int* __restrict__ sel_it,
    const float* __restrict__ ppls, int t) {
  const int tid = threadIdx.x;
  __shared__ float v[NB1 * 16];
  __shared__ int   gi[NB1 * 16];
  __shared__ float sv[256];
  __shared__ int   si[256];
  __shared__ int   q_s[16], c_s[16];
  __shared__ float val_s[16], loc_s[16];

  for (int i = tid; i < NB1 * 16; i += 256) { v[i] = pv[i]; gi[i] = pi[i]; }
  __syncthreads();

  for (int it = 0; it < 16; ++it) {
    int bs = tid;
    float bv = v[bs];
    for (int i = tid + 256; i < NB1 * 16; i += 256) {
      float x = v[i];
      if (x > bv || (x == bv && gi[i] < gi[bs])) { bv = x; bs = i; }
    }
    sv[tid] = bv; si[tid] = bs;
    __syncthreads();
    for (int o = 128; o; o >>= 1) {
      if (tid < o) {
        float x = sv[tid + o]; int s2 = si[tid + o];
        if (x > sv[tid] || (x == sv[tid] && gi[s2] < gi[si[tid]])) {
          sv[tid] = x; si[tid] = s2;
        }
      }
      __syncthreads();
    }
    if (tid == 0) {
      int slot = si[0], g = gi[slot];
      val_s[it] = sv[0];
      q_s[it] = g / VT;
      c_s[it] = g - (g / VT) * VT;
      v[slot] = NEGBIG;
      gi[slot] = 0x7fffffff;
    }
    __syncthreads();
  }

  // per-selection local logprob
  if (tid < 16) {
    int q = q_s[tid], c = c_s[tid];
    loc_s[tid] = (c < VP1)
        ? (logits[(size_t)q * NPAD + c] - logZ[q] + det_lp[q * RP1])
        : det_lp[q * RP1 + (c - V_)];
  }
  __syncthreads();

  // register-staged gathers (read everything, barrier, then write)
  float hreg[64];
  for (int i = 0; i < 64; ++i) {
    int idx = tid * 64 + i;
    int j = idx >> 10, d = idx & (D_ - 1);
    hreg[i] = hst[(size_t)q_s[j] * D_ + d];
  }
  float mreg[7]; int mc = 0;
  for (int i = tid; i < B_ * RP1; i += 256)
    mreg[mc++] = mask[q_s[i / RP1] * RP1 + (i % RP1)];
  int srg[2]; float lrg[2]; int sc = 0;
  for (int i = tid; i < t * B_; i += 256) {
    int tt = i >> 4, j = i & 15;
    srg[sc] = seq[tt * B_ + q_s[j]];
    lrg[sc] = lpb[tt * B_ + q_s[j]];
    ++sc;
  }
  __syncthreads();

  for (int i = 0; i < 64; ++i) hst[tid * 64 + i] = hreg[i];
  mc = 0;
  for (int i = tid; i < B_ * RP1; i += 256) mask[i] = mreg[mc++];
  sc = 0;
  for (int i = tid; i < t * B_; i += 256) { seq[i] = srg[sc]; lpb[i] = lrg[sc]; ++sc; }
  if (tid < 16) {
    seq[t * B_ + tid] = c_s[tid];
    lpb[t * B_ + tid] = loc_s[tid];
    lp_sum[tid] = (c_s[tid] == 0) ? -1000.0f : val_s[tid];
  }
  __syncthreads();

  // ROI class lookup, pointer-mask scatter (reference indexes ppls by NEW beam row)
  if (tid < 16) {
    int c = c_s[tid];
    int roi = c - V_ - 1;
    int it;
    if (roi >= 0) {
      int vis = (int)ppls[((size_t)tid * R_ + roi) * 6 + 4];
      it = vis + V_;
      mask[tid * RP1 + roi + 1] = 1.0f;
    } else {
      it = c;
    }
    sel_it[tid] = it;
  }
  __syncthreads();
  if (tid == 0) mask[0] = 0.0f;
}

// ---------------------------------------------------------------------------
// Attention over pooled ROI features + embed gather (one block per beam).
__global__ __launch_bounds__(256) void attention_k(
    const float* __restrict__ hst, const float* __restrict__ pool,
    const float* __restrict__ mask, const float* __restrict__ embedW,
    const int* __restrict__ sel_it, float* __restrict__ xt,
    float* __restrict__ ctx) {
  int b = blockIdx.x;
  __shared__ float sc[128];
  __shared__ float red[256];
  int wave = threadIdx.x >> 5, lane = threadIdx.x & 31;

  for (int r = wave; r < R_; r += 8) {
    float p = 0.f;
    for (int k = lane; k < D_; k += 32)
      p += hst[(size_t)b * D_ + k] * pool[((size_t)b * R_ + r) * D_ + k];
    for (int o = 16; o; o >>= 1) p += __shfl_down(p, o, 32);
    if (lane == 0) {
      float s = p * (1.0f / 32.0f);  // 1/sqrt(1024)
      if (mask[b * RP1 + 1 + r] > 0.f) s = NEGL;
      sc[r] = s;
    }
  }
  __syncthreads();

  float lm = NEGBIG;
  for (int r = threadIdx.x; r < R_; r += 256) lm = fmaxf(lm, sc[r]);
  red[threadIdx.x] = lm;
  __syncthreads();
  for (int o = 128; o; o >>= 1) {
    if (threadIdx.x < o) red[threadIdx.x] = fmaxf(red[threadIdx.x], red[threadIdx.x + o]);
    __syncthreads();
  }
  float M = red[0];
  __syncthreads();
  float ls = 0.f;
  for (int r = threadIdx.x; r < R_; r += 256) ls += __expf(sc[r] - M);
  red[threadIdx.x] = ls;
  __syncthreads();
  for (int o = 128; o; o >>= 1) {
    if (threadIdx.x < o) red[threadIdx.x] += red[threadIdx.x + o];
    __syncthreads();
  }
  float S = red[0];
  __syncthreads();
  for (int r = threadIdx.x; r < R_; r += 256) sc[r] = __expf(sc[r] - M) / S;
  __syncthreads();

  int it = sel_it[b];
  for (int d = threadIdx.x; d < D_; d += 256) {
    float acc = 0.f;
    for (int r = 0; r < R_; ++r)
      acc += sc[r] * pool[((size_t)b * R_ + r) * D_ + d];
    ctx[(size_t)b * D_ + d] = acc;
    xt[(size_t)b * D_ + d] = embedW[(size_t)it * D_ + d];
  }
}

// det head: det_lg = h_new @ Wd  (tiny, fp32)
__global__ void det_head(const float* __restrict__ hst, const float* __restrict__ Wd,
                         float* __restrict__ det_lg) {
  int b = blockIdx.x, v = threadIdx.x;
  if (v >= RP1) return;
  float acc = 0.f;
  for (int k = 0; k < D_; ++k)
    acc += hst[(size_t)b * D_ + k] * Wd[(size_t)k * RP1 + v];
  det_lg[b * RP1 + v] = acc;
}

__global__ __launch_bounds__(256) void init_state(
    const float* __restrict__ rnn_in, const float* __restrict__ det_in,
    const float* __restrict__ h_in, const float* __restrict__ mask_in,
    float* __restrict__ rnn, float* __restrict__ det_lg, float* __restrict__ hst,
    float* __restrict__ mask, float* __restrict__ lp_sum, int* __restrict__ seq,
    float* __restrict__ lpb) {
  int i = blockIdx.x * 256 + threadIdx.x;
  if (i < B_ * D_) { rnn[i] = rnn_in[i]; hst[i] = h_in[i]; }
  if (i < B_ * RP1) { det_lg[i] = det_in[i]; mask[i] = mask_in[i]; }
  if (i < B_) lp_sum[i] = 0.f;
  if (i < T_ * B_) { seq[i] = 0; lpb[i] = 0.f; }
}

__global__ __launch_bounds__(256) void finalize_k(const int* __restrict__ seq,
                                                  const float* __restrict__ lpb,
                                                  const float* __restrict__ lps,
                                                  float* __restrict__ out) {
  int i = blockIdx.x * 256 + threadIdx.x;
  if (i < T_ * B_) ((int*)out)[i] = seq[i];               // beam_seq (int32 bits)
  else if (i < 2 * T_ * B_) out[i] = lpb[i - T_ * B_];    // beam_logprobs
  else if (i < 2 * T_ * B_ + B_) out[i] = lps[i - 2 * T_ * B_];  // logprobs_sum
}

// ---------------------------------------------------------------------------
extern "C" void kernel_launch(void* const* d_in, const int* in_sizes, int n_in,
                              void* d_out, int out_size, void* d_ws, size_t ws_size,
                              hipStream_t stream) {
  const float* embed_W   = (const float*)d_in[0];
  const float* Wx        = (const float*)d_in[1];
  const float* Wh        = (const float*)d_in[2];
  const float* Wc        = (const float*)d_in[3];
  const float* Wl        = (const float*)d_in[4];
  const float* Wd        = (const float*)d_in[5];
  const float* rnn_in    = (const float*)d_in[6];
  const float* det_in    = (const float*)d_in[7];
  const float* h_in      = (const float*)d_in[8];
  const float* pool      = (const float*)d_in[9];
  const float* ppls      = (const float*)d_in[10];
  const float* pmask     = (const float*)d_in[11];
  (void)in_sizes; (void)n_in; (void)out_size; (void)ws_size;

  uint8_t* ws = (uint8_t*)d_ws;
  uint16_t* WlPack = (uint16_t*)(ws + OFF_WLP);
  uint16_t* XPack  = (uint16_t*)(ws + OFF_XP);
  uint16_t* ApW    = (uint16_t*)(ws + OFF_APW);
  uint16_t* ApX    = (uint16_t*)(ws + OFF_APX);
  float* logits = (float*)(ws + OFF_LOG);
  float* logZ   = (float*)(ws + OFF_LZ);
  float* det_lp = (float*)(ws + OFF_DLP);
  float* det_lg = (float*)(ws + OFF_DLG);
  float* mask   = (float*)(ws + OFF_MSK);
  float* rnn    = (float*)(ws + OFF_RNN);
  float* hst    = (float*)(ws + OFF_HST);
  float* xt     = (float*)(ws + OFF_XT);
  float* ctx    = (float*)(ws + OFF_CTX);
  float* lp_sum = (float*)(ws + OFF_LPS);
  int*   seq    = (int*)  (ws + OFF_SEQ);
  float* lpb    = (float*)(ws + OFF_LPB);
  float* pvb    = (float*)(ws + OFF_PV);
  int*   pib    = (int*)  (ws + OFF_PI);
  int*   sel_it = (int*)  (ws + OFF_SEL);

  // one-time repacks: Wl -> bf16 B-frag layout (102.4MB, L2-resident afterwards)
  {
    size_t n = (size_t)D_ * NPAD;
    pack_wl<<<dim3((unsigned)((n + 255) / 256)), 256, 0, stream>>>(Wl, WlPack);
  }
  {
    size_t n = (size_t)K3 * D_;
    pack_wxhc<<<dim3((unsigned)((n + 255) / 256)), 256, 0, stream>>>(Wx, Wh, Wc, XPack);
  }
  init_state<<<64, 256, 0, stream>>>(rnn_in, det_in, h_in, pmask, rnn, det_lg,
                                     hst, mask, lp_sum, seq, lpb);

  for (int t = 0; t < T_; ++t) {
    det_lsm<<<1, 32, 0, stream>>>(det_lg, det_lp);
    packA_wl<<<64, 256, 0, stream>>>(rnn, ApW);
    gemm_wl<<<(NT_WL + 7) / 8, 256, 0, stream>>>(ApW, WlPack, logits);
    lse_k<<<B_, 256, 0, stream>>>(logits, logZ);
    topk1<<<NB1, 256, 0, stream>>>(logits, logZ, det_lp, lp_sum, t, pvb, pib);
    topk2_update<<<1, 256, 0, stream>>>(pvb, pib, logits, logZ, det_lp, lp_sum,
                                        seq, lpb, hst, mask, sel_it, ppls, t);
    attention_k<<<B_, 256, 0, stream>>>(hst, pool, mask, embed_W, sel_it, xt, ctx);
    packA_x<<<(B_ * K3 + 255) / 256, 256, 0, stream>>>(xt, hst, ctx, ApX);
    gemm_x_tanh<<<NT_X / 8, 256, 0, stream>>>(ApX, XPack, rnn, hst);
    det_head<<<B_, 128, 0, stream>>>(hst, Wd, det_lg);
  }
  finalize_k<<<3, 256, 0, stream>>>(seq, lpb, lp_sum, (float*)d_out);
}
